// ResidualLSTMNetwork_8546984919123
// MI455X (gfx1250) — matrix operand order repaired and checked by reading, not compiled, not run on hardware
//
#include <hip/hip_runtime.h>

// ---------------- problem constants ----------------
#define UNITS   4
#define HID     1024     // hidden size H
#define BATCH   32       // B
#define TSTEPS  1024     // T
#define G4H     (4*HID)  // 4096
#define KTOT    (2*HID)  // concatenated K: [h ; x] = 2048
#define NWG     64       // workgroups in persistent scan (each owns 16 hidden cols)

typedef __attribute__((ext_vector_type(16))) __bf16 v16bf;
typedef __attribute__((ext_vector_type(8)))  float  v8f;

// fragment counts
#define A_FRAGS_PER_BUF (2*32*32)          // [mtile2][ktile32][lane32] (h part, K=1024)
#define HFRAG_ELEMS     (2*32*32*16)       // bf16 elems per h buffer = 32768

// ---------------------------------------------------------------------------
// Pack [U;W] (fp32, each [H,4H] row-major) into bf16 B-fragment layout:
//   out[ntile(256)][ktile(64)][lane(32)][elem(16)]
// B 32x16 16-bit layout: lanes 0-15 -> N=lane,    elems0-7 K=kb+0..7,  elems8-15 K=kb+16..23
//                        lanes16-31 -> N=lane-16, elems0-7 K=kb+8..15, elems8-15 K=kb+24..31
// ---------------------------------------------------------------------------
__global__ __launch_bounds__(256) void pack_wu_kernel(const float* __restrict__ W,
                                                      const float* __restrict__ U,
                                                      __bf16* __restrict__ outp) {
    size_t idx = (size_t)blockIdx.x * 256 + threadIdx.x;   // total 2048*4096 = 8388608
    int elem  = (int)(idx & 15);
    int lane  = (int)((idx >> 4) & 31);
    int kt    = (int)((idx >> 9) & 63);
    int ntile = (int)(idx >> 15);
    int kk = (elem & 7) | ((lane >= 16) ? 8 : 0) | ((elem >= 8) ? 16 : 0);
    int k  = kt * 32 + kk;                 // 0..2047 : 0..1023 = h (U), 1024..2047 = x (W)
    int n  = ntile * 16 + (lane & 15);     // 0..4095
    float v = (k < HID) ? U[(size_t)k * G4H + n]
                        : W[(size_t)(k - HID) * G4H + n];
    outp[idx] = (__bf16)v;
}

// ---------------------------------------------------------------------------
// Pack layer input x (fp32 [B,T,H]) into per-timestep bf16 A-fragment slabs:
//   out[t][mtile(2)][ktile(32)][lane(32)][elem(16)]
// A 16x32 16-bit layout: lane<16: row=lane,  elems0-7 K=kb+0..7,  elems8-15 K=kb+16..23
//                        lane>=16: row=l-16, elems0-7 K=kb+8..15, elems8-15 K=kb+24..31
// ---------------------------------------------------------------------------
__global__ __launch_bounds__(256) void pack_x_kernel(const float* __restrict__ x,
                                                     __bf16* __restrict__ outp) {
    size_t idx = (size_t)blockIdx.x * 256 + threadIdx.x;   // total T*2*32*32*16 = 33554432
    int elem = (int)(idx & 15);
    int lane = (int)((idx >> 4) & 31);
    int kt   = (int)((idx >> 9) & 31);
    int m    = (int)((idx >> 14) & 1);
    int t    = (int)(idx >> 15);
    int kk = (elem & 7) | ((lane >= 16) ? 8 : 0) | ((elem >= 8) ? 16 : 0);
    int k  = kt * 32 + kk;                 // hidden index 0..1023
    int b  = m * 16 + (lane & 15);         // batch row
    outp[idx] = (__bf16)x[((size_t)b * TSTEPS + t) * HID + k];
}

// zero barrier counter + both h fragment buffers (h0 = 0)
__global__ __launch_bounds__(256) void init_kernel(unsigned* __restrict__ cnt,
                                                   __bf16* __restrict__ hfrag) {
    int i = blockIdx.x * 256 + threadIdx.x;
    if (i == 0) *cnt = 0u;
    if (i < 2 * HFRAG_ELEMS) hfrag[i] = (__bf16)0.0f;
}

__device__ __forceinline__ float sigmf(float z) {
    return 1.0f / (1.0f + __expf(-z));
}

// ---------------------------------------------------------------------------
// Persistent recurrence kernel. 64 WGs x 128 threads (4 waves).
//  WG wg owns hidden cols j0=wg*16 .. j0+15. Wave g computes gate g's z-tile
//  (N cols g*1024+j0 .. +15) for both M-tiles via 128 v_wmma per step.
//  Grid barrier per timestep; h double-buffered in packed A-frag layout.
// ---------------------------------------------------------------------------
__global__ __launch_bounds__(128, 1) void lstm_scan_kernel(
        const float*  __restrict__ xin,    // [B,T,H] fp32 layer input (residual)
        const __bf16* __restrict__ wu,     // packed [U;W] B-fragments
        const __bf16* __restrict__ xfrag,  // packed x A-fragments, per-t slabs
        const float*  __restrict__ bias,   // [4H]
        float*        __restrict__ out,    // [B,T,H] fp32 layer output
        __bf16*       __restrict__ hfrag,  // 2 x packed h A-fragment buffers
        unsigned*     __restrict__ cnt) {  // grid barrier counter (monotonic)

    __shared__ float ldsZ[4][32][16];      // [gate][batch row][jj]

    const int tid  = threadIdx.x;
    const int wave = tid >> 5;             // gate id 0..3 (i,f,g,o)
    const int lane = tid & 31;
    const int wg   = blockIdx.x;
    const int j0   = wg * 16;

    // B fragments for this wave's N-tile: ntile = gate*64 + wg
    const v16bf* wuT = (const v16bf*)wu + (size_t)(wave * 64 + wg) * 64 * 32;

    // fixed per-thread elementwise mapping: jj = tid%16 constant across iters
    const int   jj = tid & 15;
    const int   j  = j0 + jj;
    const float bi = bias[0 * HID + j];
    const float bf = bias[1 * HID + j];
    const float bg = bias[2 * HID + j];
    const float bo = bias[3 * HID + j];

    // h-fragment write coordinates for column j (constant)
    const int ktileJ  = j >> 5;
    const int kkJ     = j & 31;
    const int laneOff = (kkJ & 8) ? 16 : 0;
    const int elemJ   = (kkJ & 7) | ((kkJ & 16) ? 8 : 0);

    float creg[4] = {0.f, 0.f, 0.f, 0.f};  // cell state, 4 (b,j) pairs per thread
    unsigned round = 0;

    for (int t = 0; t < TSTEPS; ++t) {
        v8f acc0 = {}, acc1 = {};
        const v16bf* hf = (const v16bf*)hfrag + (size_t)(t & 1) * A_FRAGS_PER_BUF;
        const v16bf* xf = (const v16bf*)xfrag + (size_t)t * A_FRAGS_PER_BUF;

        // ---- h @ U : K tiles 0..31 ----
        #pragma unroll 4
        for (int kt = 0; kt < 32; ++kt) {
            v16bf bfr = wuT[(size_t)kt * 32 + lane];
            v16bf a0  = hf[(size_t)(0 * 32 + kt) * 32 + lane];
            v16bf a1  = hf[(size_t)(1 * 32 + kt) * 32 + lane];
            acc0 = __builtin_amdgcn_wmma_f32_16x16x32_bf16(false, a0, false, bfr,
                                                           (short)0, acc0, false, false);
            acc1 = __builtin_amdgcn_wmma_f32_16x16x32_bf16(false, a1, false, bfr,
                                                           (short)0, acc1, false, false);
        }
        // ---- x_t @ W : K tiles 32..63 ----
        #pragma unroll 4
        for (int kt = 0; kt < 32; ++kt) {
            v16bf bfr = wuT[(size_t)(32 + kt) * 32 + lane];
            v16bf a0  = xf[(size_t)(0 * 32 + kt) * 32 + lane];
            v16bf a1  = xf[(size_t)(1 * 32 + kt) * 32 + lane];
            acc0 = __builtin_amdgcn_wmma_f32_16x16x32_bf16(false, a0, false, bfr,
                                                           (short)0, acc0, false, false);
            acc1 = __builtin_amdgcn_wmma_f32_16x16x32_bf16(false, a1, false, bfr,
                                                           (short)0, acc1, false, false);
        }

        // scatter z tiles to LDS. C/D layout: vgpr v, lane l -> (M = v + (l<16?0:8), N = l%16)
        {
            const int col   = lane & 15;
            const int rbase = (lane < 16) ? 0 : 8;
            #pragma unroll
            for (int v = 0; v < 8; ++v) {
                ldsZ[wave][rbase + v][col]      = acc0[v];
                ldsZ[wave][16 + rbase + v][col] = acc1[v];
            }
        }
        __syncthreads();

        // elementwise gates + cell/hidden update + residual tanh output
        #pragma unroll
        for (int it = 0; it < 4; ++it) {
            const int b = (tid >> 4) + it * 8;       // batch row 0..31
            float zi = ldsZ[0][b][jj] + bi;
            float zf = ldsZ[1][b][jj] + bf;
            float zg = ldsZ[2][b][jj] + bg;
            float zo = ldsZ[3][b][jj] + bo;
            float ig = sigmf(zi);
            float fg = sigmf(zf);
            float gg = tanhf(zg);
            float og = sigmf(zo);
            float c  = fg * creg[it] + ig * gg;
            creg[it] = c;
            float h  = og * tanhf(c);

            // write h (bf16) into NEXT h fragment buffer at packed position
            const int mt = b >> 4, r = b & 15;
            size_t hIdx = ((((size_t)((t + 1) & 1) * 2 + mt) * 32 + ktileJ) * 32
                           + (r + laneOff)) * 16 + elemJ;
            hfrag[hIdx] = (__bf16)h;

            // fused residual: out = tanh(h + x_in)
            size_t xIdx = ((size_t)b * TSTEPS + t) * HID + j;
            out[xIdx] = tanhf(h + xin[xIdx]);
        }

        // ---- grid-wide barrier (monotonic counter, no reset) ----
        __threadfence();            // release this WG's h writes to device scope
        __syncthreads();
        ++round;
        if (tid == 0) {
            atomicAdd(cnt, 1u);
            while (__hip_atomic_load(cnt, __ATOMIC_RELAXED,
                                     __HIP_MEMORY_SCOPE_AGENT) < round * NWG) {
                __builtin_amdgcn_s_sleep(2);
            }
        }
        __syncthreads();
        __threadfence();            // acquire: see all WGs' new h fragments
    }
}

// ---------------------------------------------------------------------------
// Host-side orchestration (graph-capture safe: only kernel launches on stream)
// ---------------------------------------------------------------------------
extern "C" void kernel_launch(void* const* d_in, const int* in_sizes, int n_in,
                              void* d_out, int out_size, void* d_ws, size_t ws_size,
                              hipStream_t stream) {
    const float* x_in = (const float*)d_in[0];   // [B,T,H]
    const float* W    = (const float*)d_in[1];   // [UNITS,H,4H]
    const float* U    = (const float*)d_in[2];   // [UNITS,H,4H]
    const float* bias = (const float*)d_in[3];   // [UNITS,4H]
    float* out = (float*)d_out;

    char* ws = (char*)d_ws;
    size_t off = 0;
    unsigned* cnt  = (unsigned*)(ws + off); off += 256;
    __bf16* hfrag  = (__bf16*)(ws + off);   off += (size_t)2 * HFRAG_ELEMS * 2;      // 128KB
    __bf16* wupk   = (__bf16*)(ws + off);   off += (size_t)KTOT * G4H * 2;           // 16MB
    __bf16* xfrag  = (__bf16*)(ws + off);   off += (size_t)TSTEPS * HFRAG_ELEMS * 2; // 64MB
    float* bufA    = (float*)(ws + off);    off += (size_t)BATCH * TSTEPS * HID * 4; // 128MB
    float* bufB    = (float*)(ws + off);                                             // 128MB

    const float* lin[UNITS]  = { x_in, bufA, bufB, bufA };
    float*       lout[UNITS] = { bufA, bufB, bufA, out  };

    for (int l = 0; l < UNITS; ++l) {
        const size_t wstride = (size_t)HID * G4H;
        pack_wu_kernel<<<(KTOT * (size_t)G4H) / 256, 256, 0, stream>>>(
            W + (size_t)l * wstride, U + (size_t)l * wstride, wupk);
        pack_x_kernel<<<((size_t)TSTEPS * HFRAG_ELEMS) / 256, 256, 0, stream>>>(
            lin[l], xfrag);
        init_kernel<<<256, 256, 0, stream>>>(cnt, hfrag);
        lstm_scan_kernel<<<NWG, 128, 0, stream>>>(
            lin[l], wupk, xfrag, bias + (size_t)l * G4H, lout[l], hfrag, cnt);
    }
}
